// HLSQoREstimator_66434554134742
// MI455X (gfx1250) — matrix operand order repaired and checked by reading, not compiled, not run on hardware
//
#include <hip/hip_runtime.h>
#include <hip/hip_bf16.h>

#define NN 200000
#define NE 1000000
#define NB 64
#define TPB 256

typedef __bf16 bf16_t;
typedef __attribute__((ext_vector_type(16))) bf16_t v16bf;
typedef __attribute__((ext_vector_type(8)))  float  v8f;
typedef __attribute__((ext_vector_type(4)))  unsigned int u32x4;

union AFrag { v16bf v; u32x4 q[2]; unsigned short u[16]; };
union CFrag { v8f v; float f[8]; };

__device__ __forceinline__ unsigned short f2bf(float f) {
  unsigned u = __float_as_uint(f);
  u += 0x7FFFu + ((u >> 16) & 1u);          // round-to-nearest-even
  return (unsigned short)(u >> 16);
}
__device__ __forceinline__ float bf1(unsigned short s) { return __uint_as_float(((unsigned)s) << 16); }
__device__ __forceinline__ float bflo(unsigned u) { return __uint_as_float(u << 16); }
__device__ __forceinline__ float bfhi(unsigned u) { return __uint_as_float(u & 0xFFFF0000u); }
__device__ __forceinline__ int   f2ord(float f) { int i = __float_as_int(f); return (i >= 0) ? i : (i ^ 0x7FFFFFFF); }
__device__ __forceinline__ float ord2f(int i) { return __int_as_float((i >= 0) ? i : (i ^ 0x7FFFFFFF)); }
__device__ __forceinline__ float gelu_exact(float x) { return 0.5f * x * (1.f + erff(x * 0.70710678118f)); }
__device__ __forceinline__ float dotbf(u32x4 a, u32x4 b) {
  float s = 0.f;
#pragma unroll
  for (int i = 0; i < 4; ++i) s += bflo(a[i]) * bflo(b[i]) + bfhi(a[i]) * bfhi(b[i]);
  return s;
}

// ---------------------------------------------------------------------------
// WMMA GEMM: C[M,N] = A[M,K] (bf16, row-major, lda) @ Bpacked (bf16, frag order)
// KT = K/32 k-tiles (templated so A-frags stay in registers).
// MODE 0: outF = c + bias (fp32)
// MODE 1: outB = bf16(c + bias)
// MODE 2: outF = sig(skip)*(c+bias) + (1-sig(skip))*xold   (in-place safe)
// ---------------------------------------------------------------------------
template <int KT, int MODE>
__global__ __launch_bounds__(256) void wmma_gemm(
    const unsigned short* __restrict__ A, int lda,
    const unsigned short* __restrict__ Bp,
    const float* __restrict__ bias,
    int M, int N,
    float* __restrict__ outF, unsigned short* __restrict__ outB, int ldo,
    const float* __restrict__ xold, const float* __restrict__ skipp)
{
  const int wave = threadIdx.x >> 5;
  const int lane = threadIdx.x & 31;
  const int mt = blockIdx.x * (blockDim.x >> 5) + wave;
  const int row0 = mt << 4;
  if (row0 >= M) return;
  const int ntiles = N >> 4;
  const int m = lane & 15;
  const int kbase = (lane & 16) ? 16 : 0;

  AFrag a[KT];
  const unsigned short* Arow = A + (size_t)(row0 + m) * lda;
#pragma unroll
  for (int kt = 0; kt < KT; ++kt) {
    a[kt].q[0] = *(const u32x4*)(Arow + kt * 32 + kbase);
    a[kt].q[1] = *(const u32x4*)(Arow + kt * 32 + 8 + kbase);
  }
  float sk = 0.f, osk = 0.f;
  if (MODE == 2) { float s = 1.f / (1.f + __expf(-skipp[0])); sk = s; osk = 1.f - s; }

  const int nhalf = (lane & 16) ? 8 : 0;
  const int nc = lane & 15;
  for (int nt = 0; nt < ntiles; ++nt) {
    CFrag c;
#pragma unroll
    for (int r = 0; r < 8; ++r) c.f[r] = 0.f;
#pragma unroll
    for (int kt = 0; kt < KT; ++kt) {
      AFrag b;
      const unsigned short* p = Bp + (((size_t)kt * ntiles + nt) * 32 + lane) * 16;
      b.q[0] = *(const u32x4*)p;
      b.q[1] = *(const u32x4*)(p + 8);
      c.v = __builtin_amdgcn_wmma_f32_16x16x32_bf16(false, a[kt].v, false, b.v,
                                                    (short)0, c.v, false, false);
    }
    const int n = (nt << 4) + nc;
    const float bn = bias ? bias[n] : 0.f;
#pragma unroll
    for (int r = 0; r < 8; ++r) {
      const int rr = row0 + nhalf + r;
      const float vv = c.f[r] + bn;
      const size_t o = (size_t)rr * ldo + n;
      if (MODE == 0) outF[o] = vv;
      else if (MODE == 1) outB[o] = f2bf(vv);
      else outF[o] = sk * vv + osk * xold[o];
    }
  }
}

// ------------------- weight packing (B-fragment lane order) -----------------
__global__ void pack_plain(const float* __restrict__ W, int K, int N,
                           unsigned short* __restrict__ dst, int ntOff, int ntTot)
{
  int t = blockIdx.x * blockDim.x + threadIdx.x;
  if (t >= K * N) return;
  int e = t & 15, lane = (t >> 4) & 31, frag = t >> 9;
  int ntf = N >> 4;
  int kt = frag / ntf, nt = frag % ntf;
  int k = kt * 32 + ((lane & 16) ? 16 : 0) + e;
  int n = (nt << 4) + (lane & 15);
  dst[(((size_t)kt * ntTot + ntOff + nt) * 32 + lane) * 16 + e] = f2bf(W[(size_t)k * N + n]);
}

// fused  F = W(64x64) @ blockdiag(R[4][16][16]);  K=N=64
__global__ void pack_fused(const float* __restrict__ W, const float* __restrict__ R,
                           unsigned short* __restrict__ dst, int ntOff, int ntTot)
{
  int t = blockIdx.x * blockDim.x + threadIdx.x;
  if (t >= 64 * 64) return;
  int e = t & 15, lane = (t >> 4) & 31, frag = t >> 9;
  int kt = frag >> 2, nt = frag & 3;
  int k = kt * 32 + ((lane & 16) ? 16 : 0) + e;
  int n = (nt << 4) + (lane & 15);
  int h = n >> 4, f = n & 15;
  float acc = 0.f;
  for (int d = 0; d < 16; ++d) acc += W[(size_t)k * 64 + h * 16 + d] * R[(h * 16 + d) * 16 + f];
  dst[(((size_t)kt * ntTot + ntOff + nt) * 32 + lane) * 16 + e] = f2bf(acc);
}

// bias192 = [bq | bk@blockdiag(Ra) | bv@blockdiag(Rm)]
__global__ void bias192_kernel(const float* __restrict__ bq, const float* __restrict__ bk,
                               const float* __restrict__ Ra, const float* __restrict__ bv,
                               const float* __restrict__ Rm, float* __restrict__ dst)
{
  int n = threadIdx.x;  // 192
  float v;
  if (n < 64) v = bq[n];
  else {
    int nn = n & 63, h = nn >> 4, f = nn & 15;
    const float* b = (n < 128) ? bk : bv;
    const float* R = (n < 128) ? Ra : Rm;
    v = 0.f;
    for (int d = 0; d < 16; ++d) v += b[h * 16 + d] * R[(h * 16 + d) * 16 + f];
  }
  dst[n] = v;
}

// ------------------------------ utility kernels -----------------------------
__global__ void fill_f32(float* p, float v, int n) { int i = blockIdx.x * blockDim.x + threadIdx.x; if (i < n) p[i] = v; }
__global__ void fill_i32(int* p, int v, int n)     { int i = blockIdx.x * blockDim.x + threadIdx.x; if (i < n) p[i] = v; }
__global__ void convert_bf(const float* __restrict__ x, unsigned short* __restrict__ o, int n)
{ int i = blockIdx.x * blockDim.x + threadIdx.x; if (i < n) o[i] = f2bf(x[i]); }
__global__ void gelu_bf_kernel(const float* __restrict__ x, unsigned short* __restrict__ o, int n)
{ int i = blockIdx.x * blockDim.x + threadIdx.x; if (i < n) o[i] = f2bf(gelu_exact(x[i])); }
__global__ void count_nodes(const int* __restrict__ batch, float* __restrict__ cnt, int n)
{ int i = blockIdx.x * blockDim.x + threadIdx.x; if (i < n) atomicAdd(&cnt[batch[i]], 1.f); }

// ------------------------------ graph LayerNorm -----------------------------
__global__ void graph_stats(const float* __restrict__ x, const int* __restrict__ batch,
                            float* __restrict__ gsum, float* __restrict__ gsq, int n)
{
  int i = blockIdx.x * blockDim.x + threadIdx.x;
  if (i >= n) return;
  const float* r = x + (size_t)i * 64;
  float s = 0.f, ss = 0.f;
#pragma unroll 8
  for (int c = 0; c < 64; ++c) { float v = r[c]; s += v; ss += v * v; }
  int b = batch[i];
  atomicAdd(&gsum[b], s);
  atomicAdd(&gsq[b], ss);
}

__global__ void graph_ln_apply(float* __restrict__ x, const int* __restrict__ batch,
                               const float* __restrict__ cnt, const float* __restrict__ gsum,
                               const float* __restrict__ gsq, const float* __restrict__ gam,
                               const float* __restrict__ bet, unsigned short* __restrict__ xb,
                               unsigned short* __restrict__ jkb, int jkoff, int n)
{
  int t = blockIdx.x * blockDim.x + threadIdx.x;
  if (t >= n * 64) return;
  int row = t >> 6, c = t & 63;
  int b = batch[row];
  float f = fmaxf(cnt[b] * 64.f, 1.f);
  float mean = gsum[b] / f;
  float var = fmaxf(gsq[b] / f - mean * mean, 0.f);
  float v = (x[t] - mean) * rsqrtf(var + 1e-5f) * gam[c] + bet[c];
  x[t] = v;
  unsigned short bv = f2bf(v);
  xb[t] = bv;
  if (jkb) jkb[(size_t)row * 192 + jkoff + c] = bv;
}

// ------------------------------- edge kernels -------------------------------
// qkv layout per node: cols [0,64)=q, [64,128)=k_rel, [128,192)=v_rel (bf16)
__global__ void edge_scores(const int* __restrict__ src, const int* __restrict__ dst,
                            const unsigned short* __restrict__ qb, const unsigned short* __restrict__ kb,
                            const float* __restrict__ prel, float* __restrict__ abuf,
                            int* __restrict__ amax, int E)
{
  int t = blockIdx.x * blockDim.x + threadIdx.x;
  if (t >= E * 4) return;
  int e = t >> 2, h = t & 3;
  int s = src[e], d = dst[e];
  const u32x4* qp = (const u32x4*)(qb + (size_t)d * 192 + h * 16);
  const u32x4* kp = (const u32x4*)(kb + (size_t)s * 192 + 64 + h * 16);
  float acc = dotbf(qp[0], kp[0]) + dotbf(qp[1], kp[1]);
  float a = acc * prel[h] * 0.25f;   // 1/sqrt(D)=0.25
  abuf[t] = a;
  atomicMax(&amax[d * 4 + h], f2ord(a));
}

__global__ void edge_exp(const int* __restrict__ dst, float* __restrict__ abuf,
                         const int* __restrict__ amax, float* __restrict__ den, int E)
{
  int t = blockIdx.x * blockDim.x + threadIdx.x;
  if (t >= E * 4) return;
  int e = t >> 2, h = t & 3;
  int d = dst[e];
  float w = __expf(abuf[t] - ord2f(amax[d * 4 + h]));
  abuf[t] = w;
  atomicAdd(&den[d * 4 + h], w);
}

__global__ void edge_scatter(const int* __restrict__ src, const int* __restrict__ dst,
                             const float* __restrict__ abuf, const float* __restrict__ den,
                             const unsigned short* __restrict__ vb, float* __restrict__ agg, int E)
{
  long long t = (long long)blockIdx.x * blockDim.x + threadIdx.x;
  if (t >= (long long)E * 64) return;
  int e = (int)(t >> 6), c = (int)(t & 63), h = c >> 4;
  int s = src[e], d = dst[e];
  float w = abuf[e * 4 + h] / fmaxf(den[d * 4 + h], 1e-16f);
  float v = bf1(vb[(size_t)s * 192 + 128 + c]);
  atomicAdd(&agg[(size_t)d * 64 + c], w * v);
}

// ------------------------------- pooling + MLPs -----------------------------
__global__ void pool_accum(const float* __restrict__ x, const int* __restrict__ batch,
                           float* __restrict__ psum, int* __restrict__ pmax, int n)
{
  int t = blockIdx.x * blockDim.x + threadIdx.x;
  if (t >= n * 64) return;
  int row = t >> 6, c = t & 63;
  int b = batch[row];
  float v = x[t];
  atomicAdd(&psum[b * 64 + c], v);
  atomicMax(&pmax[b * 64 + c], f2ord(v));
}

__global__ void ntmlp_kernel(const float* __restrict__ psum, const int* __restrict__ pmax,
                             const float* __restrict__ cnt,
                             const float* __restrict__ W1, const float* __restrict__ b1,
                             const float* __restrict__ lng, const float* __restrict__ lnb,
                             const float* __restrict__ W2, const float* __restrict__ b2,
                             float* __restrict__ xaggr, int colOff)
{
  int g = blockIdx.x, j = threadIdx.x;  // 64 threads
  __shared__ float p[192];
  __shared__ float h[64];
  __shared__ float g2[64];
  float s = psum[g * 64 + j];
  float c = fmaxf(cnt[g], 1.f);
  int mi = pmax[g * 64 + j];
  float mx = (mi == (int)0x80000000) ? 0.f : ord2f(mi);
  p[j] = s; p[64 + j] = s / c; p[128 + j] = mx;
  __syncthreads();
  float acc = b1[j];
  for (int i = 0; i < 192; ++i) acc += p[i] * W1[i * 64 + j];
  h[j] = acc; __syncthreads();
  float m = 0.f; for (int i = 0; i < 64; ++i) m += h[i]; m *= (1.f / 64.f);
  float v = 0.f; for (int i = 0; i < 64; ++i) { float d = h[i] - m; v += d * d; } v *= (1.f / 64.f);
  float hn = (acc - m) * rsqrtf(v + 1e-5f) * lng[j] + lnb[j];
  g2[j] = gelu_exact(hn); __syncthreads();
  if (j < 32) {
    float o = b2[j];
    for (int i = 0; i < 64; ++i) o += g2[i] * W2[i * 32 + j];
    xaggr[g * 80 + colOff + j] = o;
  }
}

__global__ void ymlp_kernel(const float* __restrict__ y, const float* __restrict__ W1,
                            const float* __restrict__ b1, const float* __restrict__ al,
                            const float* __restrict__ W2, const float* __restrict__ b2,
                            float* __restrict__ xaggr)
{
  int g = blockIdx.x, j = threadIdx.x;  // 16 threads
  __shared__ float h[16];
  float acc = b1[j];
  for (int i = 0; i < 4; ++i) acc += y[g * 4 + i] * W1[i * 16 + j];
  acc = (acc > 0.f) ? acc : al[j] * acc;
  h[j] = acc; __syncthreads();
  float o = b2[j];
  for (int i = 0; i < 16; ++i) o += h[i] * W2[i * 16 + j];
  xaggr[g * 80 + 64 + j] = o;
}

__global__ void gmlp_kernel(const float* __restrict__ xa, const float* __restrict__ W1,
                            const float* __restrict__ b1, const float* __restrict__ lng,
                            const float* __restrict__ lnb, const float* __restrict__ W2,
                            const float* __restrict__ b2, float* __restrict__ out)
{
  int g = blockIdx.x, j = threadIdx.x;  // 64 threads
  __shared__ float h[64];
  __shared__ float g2[64];
  float acc = b1[j];
  for (int i = 0; i < 80; ++i) acc += xa[g * 80 + i] * W1[i * 64 + j];
  h[j] = acc; __syncthreads();
  float m = 0.f; for (int i = 0; i < 64; ++i) m += h[i]; m *= (1.f / 64.f);
  float v = 0.f; for (int i = 0; i < 64; ++i) { float d = h[i] - m; v += d * d; } v *= (1.f / 64.f);
  float hn = (acc - m) * rsqrtf(v + 1e-5f) * lng[j] + lnb[j];
  g2[j] = gelu_exact(hn); __syncthreads();
  if (j < 4) {
    float o = b2[j];
    for (int i = 0; i < 64; ++i) o += g2[i] * W2[i * 4 + j];
    out[g * 4 + j] = o;
  }
}

// ---------------------------------------------------------------------------
extern "C" void kernel_launch(void* const* d_in, const int* in_sizes, int n_in,
                              void* d_out, int out_size, void* d_ws, size_t ws_size,
                              hipStream_t stream)
{
  (void)in_sizes; (void)n_in; (void)out_size; (void)ws_size;
  const float* x_in[2] = { (const float*)d_in[0], (const float*)d_in[1] };
  const int* ei[2]     = { (const int*)d_in[2], (const int*)d_in[3] };
  const int* batch[2]  = { (const int*)d_in[4], (const int*)d_in[5] };
  const float* ybase   = (const float*)d_in[6];
  auto P = [&](int i) -> const float* { return (const float*)d_in[7 + i]; };
  // param leaf order (jax pytree, sorted dict keys):
  // 0..5 gmlp{l1.b,l1.w,l2.b,l2.w,ln.b,ln.g}; 6..9 jk_out{op.b,op.w,var.b,var.w};
  // 10+28*l layers: a_rel[ov,vo], k{op.b,op.w,var.b,var.w}, m_rel[ov,vo],
  //   norm{op.b,op.g,var.b,var.g}, out{..}, p_rel[ov,vo], q{..}, skip[op,var], v{..}
  // 94..105 ntmlp op/var {l1.b,l1.w,l2.b,l2.w,ln.b,ln.g}
  // 106..109 proj; 110..113 proj_ln; 114..118 ymlp{l1.b,l1.w,l2.b,l2.w,prelu}

  char* base = (char*)d_ws; size_t off = 0;
  auto alloc = [&](size_t b) -> void* { off = (off + 255) & ~(size_t)255; void* p = base + off; off += b; return p; };

  float*          xf[2];  unsigned short* xb[2];  unsigned short* qkv[2];
  float*          agg[2]; unsigned short* gel[2]; unsigned short* jkb[2];
  int*            amaxI[2]; float* den[2]; float* cnt[2];
  for (int t = 0; t < 2; ++t) {
    xf[t]    = (float*)alloc((size_t)NN * 64 * 4);
    xb[t]    = (unsigned short*)alloc((size_t)NN * 64 * 2);
    qkv[t]   = (unsigned short*)alloc((size_t)NN * 192 * 2);
    agg[t]   = (float*)alloc((size_t)NN * 64 * 4);
    gel[t]   = (unsigned short*)alloc((size_t)NN * 64 * 2);
    jkb[t]   = (unsigned short*)alloc((size_t)NN * 192 * 2);
    amaxI[t] = (int*)alloc((size_t)NN * 4 * 4);
    den[t]   = (float*)alloc((size_t)NN * 4 * 4);
    cnt[t]   = (float*)alloc(NB * 4);
  }
  float* wbuf[2] = { (float*)alloc((size_t)NE * 4 * 4), (float*)alloc((size_t)NE * 4 * 4) };
  unsigned short* packB = (unsigned short*)alloc((size_t)6 * 12 * 512 * 2);
  float* bias192 = (float*)alloc(192 * 4);
  float* gsum = (float*)alloc(NB * 4);
  float* gsq  = (float*)alloc(NB * 4);
  float* psum = (float*)alloc(NB * 64 * 4);
  int*   pmaxI = (int*)alloc(NB * 64 * 4);
  float* xaggr = (float*)alloc(NB * 80 * 4);

  auto cdiv = [](long a, long b) { return (int)((a + b - 1) / b); };
  const int GN = cdiv((long)NN * 64, TPB);           // elementwise over node features
  const int GT = cdiv(NN / 16, TPB / 32);            // GEMM: 8 row-tiles per block
  const int NEG = 0x80000000;

  // ---- per-graph node counts (fixed per call, recomputed for determinism)
  for (int t = 0; t < 2; ++t) {
    fill_f32<<<1, NB, 0, stream>>>(cnt[t], 0.f, NB);
    count_nodes<<<cdiv(NN, TPB), TPB, 0, stream>>>(batch[t], cnt[t], NN);
  }

  // ---- input projection + graph LN
  for (int t = 0; t < 2; ++t) {
    convert_bf<<<GN, TPB, 0, stream>>>(x_in[t], xb[t], NN * 64);
    pack_plain<<<16, TPB, 0, stream>>>(P(107 + 2 * t), 64, 64, packB, 0, 4);
    wmma_gemm<2, 0><<<GT, TPB, 0, stream>>>(xb[t], 64, packB, P(106 + 2 * t), NN, 64,
                                            xf[t], nullptr, 64, nullptr, nullptr);
    fill_f32<<<1, NB, 0, stream>>>(gsum, 0.f, NB);
    fill_f32<<<1, NB, 0, stream>>>(gsq, 0.f, NB);
    graph_stats<<<cdiv(NN, TPB), TPB, 0, stream>>>(xf[t], batch[t], gsum, gsq, NN);
    graph_ln_apply<<<GN, TPB, 0, stream>>>(xf[t], batch[t], cnt[t], gsum, gsq,
                                           P(111 + 2 * t), P(110 + 2 * t), xb[t], nullptr, 0, NN);
  }

  // ---- HGT layers
  for (int l = 0; l < 3; ++l) {
    const int LB = 10 + 28 * l;
    for (int t = 0; t < 2; ++t) {
      const int et = t;  // op sources op__to__var (et=0); var sources var__to__op (et=1)
      // fused QKV weights -> one [64,192] GEMM (relation matrices folded in)
      pack_plain<<<16, TPB, 0, stream>>>(P(LB + 19 + 2 * t), 64, 64, packB, 0, 12);   // q
      pack_fused<<<16, TPB, 0, stream>>>(P(LB + 3 + 2 * t), P(LB + et), packB, 4, 12); // k@a_rel
      pack_fused<<<16, TPB, 0, stream>>>(P(LB + 25 + 2 * t), P(LB + 6 + et), packB, 8, 12); // v@m_rel
      bias192_kernel<<<1, 192, 0, stream>>>(P(LB + 18 + 2 * t), P(LB + 2 + 2 * t), P(LB + et),
                                            P(LB + 24 + 2 * t), P(LB + 6 + et), bias192);
      wmma_gemm<2, 1><<<GT, TPB, 0, stream>>>(xb[t], 64, packB, bias192, NN, 192,
                                              nullptr, qkv[t], 192, nullptr, nullptr);
      fill_i32<<<cdiv(NN * 4, TPB), TPB, 0, stream>>>(amaxI[t], NEG, NN * 4);
      fill_f32<<<cdiv(NN * 4, TPB), TPB, 0, stream>>>(den[t], 0.f, NN * 4);
      fill_f32<<<GN, TPB, 0, stream>>>(agg[t], 0.f, NN * 64);
    }
    for (int et = 0; et < 2; ++et) {
      const int st = et, dt = 1 - et;
      const int* src = ei[et];
      const int* dst = ei[et] + NE;
      edge_scores<<<cdiv((long)NE * 4, TPB), TPB, 0, stream>>>(src, dst, qkv[dt], qkv[st],
                                                               P(LB + 16 + et), wbuf[et], amaxI[dt], NE);
      edge_exp<<<cdiv((long)NE * 4, TPB), TPB, 0, stream>>>(dst, wbuf[et], amaxI[dt], den[dt], NE);
      edge_scatter<<<cdiv((long)NE * 64, TPB), TPB, 0, stream>>>(src, dst, wbuf[et], den[dt],
                                                                 qkv[st], agg[dt], NE);
    }
    for (int t = 0; t < 2; ++t) {
      gelu_bf_kernel<<<GN, TPB, 0, stream>>>(agg[t], gel[t], NN * 64);
      pack_plain<<<16, TPB, 0, stream>>>(P(LB + 13 + 2 * t), 64, 64, packB, 0, 4);
      wmma_gemm<2, 2><<<GT, TPB, 0, stream>>>(gel[t], 64, packB, P(LB + 12 + 2 * t), NN, 64,
                                              xf[t], nullptr, 64, xf[t], P(LB + 22 + t));
      fill_f32<<<1, NB, 0, stream>>>(gsum, 0.f, NB);
      fill_f32<<<1, NB, 0, stream>>>(gsq, 0.f, NB);
      graph_stats<<<cdiv(NN, TPB), TPB, 0, stream>>>(xf[t], batch[t], gsum, gsq, NN);
      graph_ln_apply<<<GN, TPB, 0, stream>>>(xf[t], batch[t], cnt[t], gsum, gsq,
                                             P(LB + 9 + 2 * t), P(LB + 8 + 2 * t),
                                             xb[t], jkb[t], l * 64, NN);
    }
  }

  // ---- JumpingKnowledge projection + pooling + per-type MLPs
  for (int t = 0; t < 2; ++t) {
    pack_plain<<<cdiv(192 * 64, TPB), TPB, 0, stream>>>(P(7 + 2 * t), 192, 64, packB, 0, 4);
    wmma_gemm<6, 0><<<GT, TPB, 0, stream>>>(jkb[t], 192, packB, P(6 + 2 * t), NN, 64,
                                            agg[t], nullptr, 64, nullptr, nullptr);  // agg reused as xfin
    fill_f32<<<cdiv(NB * 64, TPB), TPB, 0, stream>>>(psum, 0.f, NB * 64);
    fill_i32<<<cdiv(NB * 64, TPB), TPB, 0, stream>>>(pmaxI, NEG, NB * 64);
    pool_accum<<<GN, TPB, 0, stream>>>(agg[t], batch[t], psum, pmaxI, NN);
    const int nb = 94 + 6 * t;
    ntmlp_kernel<<<NB, 64, 0, stream>>>(psum, pmaxI, cnt[t], P(nb + 1), P(nb + 0),
                                        P(nb + 5), P(nb + 4), P(nb + 3), P(nb + 2),
                                        xaggr, t * 32);
  }
  ymlp_kernel<<<NB, 16, 0, stream>>>(ybase, P(115), P(114), P(118), P(117), P(116), xaggr);
  gmlp_kernel<<<NB, 64, 0, stream>>>(xaggr, P(1), P(0), P(5), P(4), P(3), P(2), (float*)d_out);
}
